// CustomLSTMModel_16716012716298
// MI455X (gfx1250) — matrix operand order repaired and checked by reading, not compiled, not run on hardware
//
#include <hip/hip_runtime.h>

typedef __attribute__((ext_vector_type(16))) _Float16 v16h;
typedef __attribute__((ext_vector_type(8)))  _Float16 v8h;
typedef __attribute__((ext_vector_type(8)))  float    v8f;

namespace {
constexpr int B = 64, S = 512, D = 256, H = 512;
constexpr int G = 4 * H;          // 2048 gate columns
constexpr int NT = G / 16;        // 128 N-tiles
constexpr int KT_X = D / 32;      // 8  K-steps for input GEMM
constexpr int KT_H = H / 32;      // 16 K-steps for recurrent GEMM

// workspace layout (bytes)
constexpr size_t OFF_GX   = 0;                               // fp32 [S,B,G]   256 MiB
constexpr size_t OFF_XH   = OFF_GX   + (size_t)S*B*G*4;      // f16  [B*S,D]    16 MiB
constexpr size_t OFF_WXPK = OFF_XH   + (size_t)B*S*D*2;      // f16 packed B tiles, 1 MiB
constexpr size_t OFF_UPK  = OFF_WXPK + (size_t)D*G*2;        // f16 packed B tiles, 2 MiB
constexpr size_t OFF_BX   = OFF_UPK  + (size_t)H*G*2;        // fp32 [G]
constexpr size_t OFF_HS   = OFF_BX   + (size_t)G*4;          // f16 [S,B,H]     32 MiB
constexpr size_t OFF_HB   = OFF_HS   + (size_t)S*B*H*2;      // fp32 [B,H]
constexpr size_t OFF_CB   = OFF_HB   + (size_t)B*H*4;        // fp32 [B,H]
}

// gfx1250 has a hardware V_TANH_F32 transcendental; use it for the serialized
// gate tail. sigmoid(x) == 0.5*tanh(x/2)+0.5 exactly.
#if __has_builtin(__builtin_amdgcn_tanhf)
__device__ __forceinline__ float fast_tanh(float x) { return __builtin_amdgcn_tanhf(x); }
#elif __has_builtin(__builtin_amdgcn_tanh_f32)
__device__ __forceinline__ float fast_tanh(float x) { return __builtin_amdgcn_tanh_f32(x); }
#else
__device__ __forceinline__ float fast_tanh(float x) { return tanhf(x); }
#endif
__device__ __forceinline__ float sigf(float x) { return __builtin_fmaf(0.5f, fast_tanh(0.5f * x), 0.5f); }

// Build 16x32 f16 A-fragment for one lane from row-major A.
// ISA layout: elements 0..7 -> k = kb+e, elements 8..15 -> k = 16+kb+(e-8),
// with kb = (lane>=16)*8. Two contiguous 16-byte loads.
__device__ __forceinline__ v16h load_a_frag(const _Float16* p) {
  v8h lo = *(const v8h*)(p);
  v8h hi = *(const v8h*)(p + 16);
  return __builtin_shufflevector(lo, hi, 0,1,2,3,4,5,6,7,8,9,10,11,12,13,14,15);
}

__device__ __forceinline__ v8f wmma_f16(v16h a, v16h b, v8f c) {
  return __builtin_amdgcn_wmma_f32_16x16x32_f16(false, a, false, b, (short)0, c, false, false);
}

// ---------------- pack kernels ----------------
__global__ void pack_x(const float* __restrict__ x, _Float16* __restrict__ xh) {
  size_t i = (size_t)blockIdx.x * 256 + threadIdx.x;     // B*S*D threads
  xh[i] = (_Float16)x[i];
}

// Pack W (concat along N of 4 [K,H] row-major matrices) into B-fragment tile order:
// packed[((kt*NT + nt)*32 + lane)*16 + e] = W[kt*32 + (e>>3)*16 + (lane>>4)*8 + (e&7), nt*16 + (lane&15)]
__global__ void pack_wx(const float* __restrict__ Wf, const float* __restrict__ Wi,
                        const float* __restrict__ Wo, const float* __restrict__ Wc,
                        _Float16* __restrict__ pk) {
  int p = blockIdx.x * 256 + threadIdx.x;                // D*G threads
  int e = p & 15, lane = (p >> 4) & 31, tile = p >> 9;
  int nt = tile & (NT - 1), kt = tile >> 7;
  int n = nt * 16 + (lane & 15);
  int k = kt * 32 + (e >> 3) * 16 + (lane >> 4) * 8 + (e & 7);
  int gate = n >> 9, col = n & (H - 1);
  const float* Wg = gate == 0 ? Wf : gate == 1 ? Wi : gate == 2 ? Wo : Wc;
  pk[p] = (_Float16)Wg[(size_t)k * H + col];
}

__global__ void pack_uh(const float* __restrict__ Uf, const float* __restrict__ Ui,
                        const float* __restrict__ Uo, const float* __restrict__ Uc,
                        _Float16* __restrict__ pk) {
  int p = blockIdx.x * 256 + threadIdx.x;                // H*G threads
  int e = p & 15, lane = (p >> 4) & 31, tile = p >> 9;
  int nt = tile & (NT - 1), kt = tile >> 7;
  int n = nt * 16 + (lane & 15);
  int k = kt * 32 + (e >> 3) * 16 + (lane >> 4) * 8 + (e & 7);
  int gate = n >> 9, col = n & (H - 1);
  const float* Ug = gate == 0 ? Uf : gate == 1 ? Ui : gate == 2 ? Uo : Uc;
  pk[p] = (_Float16)Ug[(size_t)k * H + col];
}

__global__ void pack_bx(const float* __restrict__ bf, const float* __restrict__ bi,
                        const float* __restrict__ bo, const float* __restrict__ bc,
                        float* __restrict__ bx) {
  int n = blockIdx.x * 256 + threadIdx.x;                // G threads
  int gate = n >> 9, col = n & (H - 1);
  const float* bg = gate == 0 ? bf : gate == 1 ? bi : gate == 2 ? bo : bc;
  bx[n] = bg[col];
}

// ---------------- input-side GEMM: gx[S,B,G] = X @ Wx + bx ----------------
// 2x2 register blocking: each wave owns a 32x32 output block (four 16x16 tiles),
// reusing each A fragment across 2 N-tiles and each B fragment across 2 M-tiles.
__global__ __launch_bounds__(256) void gemm_gx(const _Float16* __restrict__ Xh,
                                               const _Float16* __restrict__ Wpk,
                                               const float* __restrict__ bx,
                                               float* __restrict__ gx) {
  int lane = threadIdx.x & 31;
  int id = blockIdx.x * 8 + (threadIdx.x >> 5);          // one wave = one 32x32 block
  int mb = id >> 6;                                       // 1024 M-blocks (32 rows each)
  int nb = id & 63;                                       // 64  N-blocks (2 tiles each)
  int half = lane >> 4, lr = lane & 15;
  int nt0 = nb * 2, nt1 = nb * 2 + 1;
  v8f acc00 = {}, acc01 = {}, acc10 = {}, acc11 = {};
  const _Float16* arow0 = Xh + (size_t)(mb * 32 + lr) * D + half * 8;
  const _Float16* arow1 = arow0 + (size_t)16 * D;
  for (int kt = 0; kt < KT_X; ++kt) {
    v16h a0 = load_a_frag(arow0 + kt * 32);
    v16h a1 = load_a_frag(arow1 + kt * 32);
    v16h b0 = *(const v16h*)(Wpk + (((size_t)kt * NT + nt0) * 32 + lane) * 16);
    v16h b1 = *(const v16h*)(Wpk + (((size_t)kt * NT + nt1) * 32 + lane) * 16);
    acc00 = wmma_f16(a0, b0, acc00);
    acc01 = wmma_f16(a0, b1, acc01);
    acc10 = wmma_f16(a1, b0, acc10);
    acc11 = wmma_f16(a1, b1, acc11);
  }
  int col0 = nt0 * 16 + lr, col1 = nt1 * 16 + lr;
  float bias0 = bx[col0], bias1 = bx[col1];
  for (int r = 0; r < 8; ++r) {
    int m0 = mb * 32 + half * 8 + r;                      // flat b*S+s
    int m1 = m0 + 16;
    size_t o0 = ((size_t)(m0 & (S - 1)) * B + (m0 >> 9)) * G;  // [S,B,G]
    size_t o1 = ((size_t)(m1 & (S - 1)) * B + (m1 >> 9)) * G;
    gx[o0 + col0] = acc00[r] + bias0;
    gx[o0 + col1] = acc01[r] + bias1;
    gx[o1 + col0] = acc10[r] + bias0;
    gx[o1 + col1] = acc11[r] + bias1;
  }
}

// ---------------- one recurrence step ----------------
// 128 waves; wave owns (mtile in 0..3, nslice in 0..31) -> all 4 gates for a 16x16 block.
__global__ __launch_bounds__(256) void lstm_step(const float* __restrict__ gx,
                                                 const _Float16* __restrict__ Upk,
                                                 _Float16* __restrict__ hs,
                                                 float* __restrict__ hb,
                                                 float* __restrict__ cb, int t) {
  int lane = threadIdx.x & 31;
  int w = blockIdx.x * 8 + (threadIdx.x >> 5);            // 0..127
  int mtile = w & 3;
  int ns = w >> 2;                                        // 0..31
  int half = lane >> 4, lr = lane & 15;
  v8f a0 = {}, a1 = {}, a2 = {}, a3 = {};
  if (t > 0) {                                            // uniform branch: EXEC stays full
    const _Float16* arow = hs + ((size_t)(t - 1) * B + mtile * 16 + lr) * H + half * 8;
    for (int kt = 0; kt < KT_H; ++kt) {
      v16h a = load_a_frag(arow + kt * 32);
      v16h b0 = *(const v16h*)(Upk + (((size_t)kt * NT + 0 * 32 + ns) * 32 + lane) * 16);
      v16h b1 = *(const v16h*)(Upk + (((size_t)kt * NT + 1 * 32 + ns) * 32 + lane) * 16);
      v16h b2 = *(const v16h*)(Upk + (((size_t)kt * NT + 2 * 32 + ns) * 32 + lane) * 16);
      v16h b3 = *(const v16h*)(Upk + (((size_t)kt * NT + 3 * 32 + ns) * 32 + lane) * 16);
      a0 = wmma_f16(a, b0, a0);
      a1 = wmma_f16(a, b1, a1);
      a2 = wmma_f16(a, b2, a2);
      a3 = wmma_f16(a, b3, a3);
    }
  }
  int col = ns * 16 + lr;
  for (int r = 0; r < 8; ++r) {
    int bb = mtile * 16 + half * 8 + r;
    size_t gb = ((size_t)t * B + bb) * G + col;
    float fg = sigf(a0[r] + gx[gb]);
    float ig = sigf(a1[r] + gx[gb + H]);
    float og = sigf(a2[r] + gx[gb + 2 * H]);
    float cg = fast_tanh(a3[r] + gx[gb + 3 * H]);
    size_t hc = (size_t)bb * H + col;
    float cold = (t > 0) ? cb[hc] : 0.0f;
    float cn = fg * cold + ig * cg;
    cb[hc] = cn;
    float hn = og * fast_tanh(cn);
    hb[hc] = hn;
    hs[((size_t)t * B + bb) * H + col] = (_Float16)hn;
  }
}

// ---------------- output head: pred[b,s] = h[b,s,:] . Wfc + bfc ----------------
__global__ __launch_bounds__(256) void pred_head(const _Float16* __restrict__ hs,
                                                 const float* __restrict__ Wfc,
                                                 const float* __restrict__ bfc,
                                                 float* __restrict__ out) {
  int lane = threadIdx.x & 31;
  int q = blockIdx.x * 8 + (threadIdx.x >> 5);            // 0..B*S-1
  int ss = q >> 6, bb = q & (B - 1);
  v16h hv = *(const v16h*)(hs + ((size_t)ss * B + bb) * H + lane * 16);
  float sum = 0.0f;
  for (int j = 0; j < 16; ++j) sum += (float)hv[j] * Wfc[lane * 16 + j];
  for (int off = 16; off > 0; off >>= 1) sum += __shfl_xor(sum, off, 32);
  if (lane == 0) out[(size_t)bb * S + ss] = sum + bfc[0];
}

__global__ void copy_hc(const float* __restrict__ hb, const float* __restrict__ cb,
                        float* __restrict__ out) {
  int i = blockIdx.x * 256 + threadIdx.x;                 // B*H threads
  out[B * S + i] = hb[i];
  out[2 * B * S + i] = cb[i];
}

extern "C" void kernel_launch(void* const* d_in, const int* in_sizes, int n_in,
                              void* d_out, int out_size, void* d_ws, size_t ws_size,
                              hipStream_t stream) {
  (void)in_sizes; (void)n_in; (void)out_size; (void)ws_size;
  const float* x   = (const float*)d_in[0];
  const float* Wf  = (const float*)d_in[1];
  const float* Wi  = (const float*)d_in[2];
  const float* Wo  = (const float*)d_in[3];
  const float* Wc  = (const float*)d_in[4];
  const float* bf  = (const float*)d_in[5];
  const float* bi  = (const float*)d_in[6];
  const float* bo  = (const float*)d_in[7];
  const float* bc  = (const float*)d_in[8];
  const float* Uf  = (const float*)d_in[9];
  const float* Ui  = (const float*)d_in[10];
  const float* Uo  = (const float*)d_in[11];
  const float* Uc  = (const float*)d_in[12];
  const float* Wfc = (const float*)d_in[13];
  const float* bfc = (const float*)d_in[14];

  char* ws = (char*)d_ws;
  float*    gx  = (float*)   (ws + OFF_GX);
  _Float16* xh  = (_Float16*)(ws + OFF_XH);
  _Float16* wpk = (_Float16*)(ws + OFF_WXPK);
  _Float16* upk = (_Float16*)(ws + OFF_UPK);
  float*    bx  = (float*)   (ws + OFF_BX);
  _Float16* hs  = (_Float16*)(ws + OFF_HS);
  float*    hb  = (float*)   (ws + OFF_HB);
  float*    cb  = (float*)   (ws + OFF_CB);
  float*    out = (float*)d_out;

  pack_x  <<<(B * S * D) / 256, 256, 0, stream>>>(x, xh);
  pack_wx <<<(D * G) / 256,     256, 0, stream>>>(Wf, Wi, Wo, Wc, wpk);
  pack_uh <<<(H * G) / 256,     256, 0, stream>>>(Uf, Ui, Uo, Uc, upk);
  pack_bx <<<G / 256,           256, 0, stream>>>(bf, bi, bo, bc, bx);

  // 1024 M-blocks x 64 N-blocks (32x32 per wave), 8 waves per 256-thread block
  gemm_gx <<<(1024 * 64) / 8, 256, 0, stream>>>(xh, wpk, bx, gx);

  for (int t = 0; t < S; ++t)
    lstm_step<<<16, 256, 0, stream>>>(gx, upk, hs, hb, cb, t);

  pred_head<<<(B * S) / 8, 256, 0, stream>>>(hs, Wfc, bfc, out);
  copy_hc  <<<(B * H) / 256, 256, 0, stream>>>(hb, cb, out);
}
